// Encoder_15247133901613
// MI455X (gfx1250) — matrix operand order repaired and checked by reading, not compiled, not run on hardware
//
#include <hip/hip_runtime.h>
#include <hip/hip_bf16.h>
#include <math.h>

// Problem dims (fixed by reference)
#define B_ 512
#define T_ 128
#define F_ 128
#define H_ 256
#define BT (B_ * T_)
#define G4H (4 * H_)   // 1024
#define K2F (2 * F_)   // 256

typedef __attribute__((ext_vector_type(16))) _Float16 v16h;
typedef __attribute__((ext_vector_type(8)))  float    v8f;

// ---------------------------------------------------------------------------
// WMMA operand loaders (CDNA5 wave32, 16x16x32 f16 -> f32), contiguous K order, same in A and B:
//  A (16x32, row-major source, row = M):
//    lane L: m = L%16, half = L>>4
//    dword d (0..7): k = half*16 + d*2, holds {k, k+1}
//  B (32x16, from row-major [N,K] weight, so out = X @ W^T):
//    lane L: n = L%16, half = L>>4
//    element i (0..15): k = half*16 + i   (contiguous run of 16 K per lane)
// ---------------------------------------------------------------------------
__device__ __forceinline__ v16h ld_a(const _Float16* p, int ld) {
  const int lane = threadIdx.x & 31;
  const int row  = lane & 15;
  const int half = lane >> 4;
  const _Float16* rp = p + row * ld;
  v16h r;
#pragma unroll
  for (int d = 0; d < 8; ++d) {
    const int k = (half << 4) + (d << 1);
    r[2 * d]     = rp[k];
    r[2 * d + 1] = rp[k + 1];
  }
  return r;
}

__device__ __forceinline__ v16h ld_b(const _Float16* p, int ld) {
  const int lane = threadIdx.x & 31;
  const int col  = lane & 15;   // N
  const int half = lane >> 4;   // K half (0: K=0..15, 1: K=16..31)
  const _Float16* rp = p + col * ld + half * 16;
  v16h r;
#pragma unroll
  for (int i = 0; i < 16; ++i) r[i] = rp[i];
  return r;
}

__device__ __forceinline__ v8f wmma16(v16h a, v16h b, v8f c) {
  // (neg_a, A, neg_b, B, c_mod, C, reuse_a, reuse_b)
  return __builtin_amdgcn_wmma_f32_16x16x32_f16(false, a, false, b, (short)0, c,
                                                false, false);
}

__device__ __forceinline__ float sigf(float x) { return 1.f / (1.f + __expf(-x)); }

// ---------------------------------------------------------------------------
// Staging kernels: f32 -> f16 (and masked / transposed variants)
// ---------------------------------------------------------------------------
__global__ void k_cast_f16(const float* __restrict__ s, _Float16* __restrict__ d,
                           int n) {
  int i = blockIdx.x * 256 + threadIdx.x;
  if (i < n) d[i] = (_Float16)s[i];
}

__global__ void k_cast_nodiag(const float* __restrict__ s,
                              _Float16* __restrict__ d, int dim) {
  int i = blockIdx.x * 256 + threadIdx.x;
  if (i < dim * dim) {
    int r = i / dim, c = i % dim;
    d[i] = (r == c) ? (_Float16)0.f : (_Float16)s[i];
  }
}

// values [B,T,F] f32 -> valsT [B,F,T] f16 (for temporal GEMM B-operand)
__global__ void k_transpose_f16(const float* __restrict__ s,
                                _Float16* __restrict__ d) {
  int i = blockIdx.x * 256 + threadIdx.x;
  if (i < B_ * T_ * F_) {
    int f = i % F_;
    int t = (i / F_) % T_;
    int b = i / (F_ * T_);
    d[((long)b * F_ + f) * T_ + t] = (_Float16)s[i];
  }
}

// ---------------------------------------------------------------------------
// Fused imputation: vhat = V @ (feat_W*(1-I))^T + feat_b
//                        + (temp_W*(1-I)) @ V_b  + temp_b   (per batch b)
//                   XH   = V*M + vhat*(1-M)            -> f16 [B,T,F]
// grid (T/16, B), 256 thr; wave w owns f-tile w*16.
// ---------------------------------------------------------------------------
__global__ __launch_bounds__(256) void k_impute(
    const _Float16* __restrict__ vals16, const _Float16* __restrict__ valsT16,
    const _Float16* __restrict__ featW16, const _Float16* __restrict__ tempW16,
    const float* __restrict__ values, const float* __restrict__ masks,
    const float* __restrict__ feat_b, const float* __restrict__ temp_b,
    _Float16* __restrict__ XH16) {
  const int b  = blockIdx.y;
  const int t0 = blockIdx.x * 16;
  const int w  = threadIdx.x >> 5;
  const int f0 = w * 16;

  v8f acc = {};
  // Feature regression: rows = (b,t), K over source feature f
#pragma unroll
  for (int kk = 0; kk < F_ / 32; ++kk) {
    v16h a  = ld_a(vals16 + ((long)b * T_ + t0) * F_ + kk * 32, F_);
    v16h wb = ld_b(featW16 + (long)f0 * F_ + kk * 32, F_);
    acc = wmma16(a, wb, acc);
  }
  // Temporal regression: out[t,f] += sum_t' tempW[t,t'] * V[b,t',f]
#pragma unroll
  for (int kk = 0; kk < T_ / 32; ++kk) {
    v16h a  = ld_a(tempW16 + (long)t0 * T_ + kk * 32, T_);
    v16h wb = ld_b(valsT16 + ((long)b * F_ + f0) * T_ + kk * 32, T_);
    acc = wmma16(a, wb, acc);
  }

  const int lane = threadIdx.x & 31;
  const int n    = lane & 15;
  const int mb   = (lane >> 4) << 3;
  const float fb = feat_b[f0 + n];
#pragma unroll
  for (int v = 0; v < 8; ++v) {
    const int t = t0 + mb + v;
    const float vhat = acc[v] + fb + temp_b[t];
    const long idx = ((long)b * T_ + t) * F_ + f0 + n;
    const float val = values[idx], m = masks[idx];
    XH16[idx] = (_Float16)(val * m + vhat * (1.f - m));
  }
}

// ---------------------------------------------------------------------------
// gamma[b,t,h] = exp(-|deltas @ decay_W^T + decay_b|)  -> f32 [BT,H]
// grid (BT/16, 2), wave w owns n-tile blockIdx.y*128 + w*16.
// ---------------------------------------------------------------------------
__global__ __launch_bounds__(256) void k_gamma(
    const _Float16* __restrict__ deltas16, const _Float16* __restrict__ decayW16,
    const float* __restrict__ decay_b, float* __restrict__ gamma) {
  const int r0 = blockIdx.x * 16;
  const int w  = threadIdx.x >> 5;
  const int n0 = blockIdx.y * 128 + w * 16;

  v8f acc = {};
#pragma unroll
  for (int kk = 0; kk < F_ / 32; ++kk) {
    v16h a  = ld_a(deltas16 + (long)r0 * F_ + kk * 32, F_);
    v16h wb = ld_b(decayW16 + (long)n0 * F_ + kk * 32, F_);
    acc = wmma16(a, wb, acc);
  }
  const int lane = threadIdx.x & 31;
  const int n    = n0 + (lane & 15);
  const int mb   = (lane >> 4) << 3;
  const float db = decay_b[n];
#pragma unroll
  for (int v = 0; v < 8; ++v)
    gamma[(long)(r0 + mb + v) * H_ + n] = __expf(-fabsf(acc[v] + db));
}

// ---------------------------------------------------------------------------
// invden[t] = 1 / (sum_{b,f} masks[b,t,f] + 1e-5), t = 1..T-1
// ---------------------------------------------------------------------------
__global__ void k_invden(const float* __restrict__ masks,
                         float* __restrict__ invden) {
  const int t = blockIdx.x + 1;
  __shared__ float red[256];
  float s = 0.f;
  for (int i = threadIdx.x; i < B_ * F_; i += 256) {
    const int b = i / F_, f = i % F_;
    s += masks[((long)b * T_ + t) * F_ + f];
  }
  red[threadIdx.x] = s;
  __syncthreads();
  for (int off = 128; off > 0; off >>= 1) {
    if (threadIdx.x < off) red[threadIdx.x] += red[threadIdx.x + off];
    __syncthreads();
  }
  if (threadIdx.x == 0) invden[t] = 1.f / (red[0] + 1e-5f);
}

// ---------------------------------------------------------------------------
// LSTM scan: one block per 16 batch rows, runs all 127 steps.
// h,c kept in f32 registers (16 elems/thread, fixed (row,col) ownership);
// f16 copies of h and [xh,m] in LDS feed the WMMA A-operands.
// A tiles are hoisted into registers once per step (16 x v16h); each wave then
// streams only B (weight) tiles from L2 for its 8 gate n-tiles.
// Loss accumulated per-thread, reduced per-block -> lossPart (deterministic).
// ---------------------------------------------------------------------------
__global__ __launch_bounds__(256) void k_scan(
    const _Float16* __restrict__ XH16, const _Float16* __restrict__ masks16,
    const float* __restrict__ gamma, const _Float16* __restrict__ Wih16,
    const _Float16* __restrict__ Whh16, const _Float16* __restrict__ outW16,
    const float* __restrict__ b_ih, const float* __restrict__ b_hh,
    const float* __restrict__ out_b, const float* __restrict__ values,
    const float* __restrict__ masks, const float* __restrict__ invden,
    float* __restrict__ h_out, float* __restrict__ c_out,
    float* __restrict__ lossPart) {
  extern __shared__ char smem[];
  float*    zbuf = (float*)smem;               // [16][1024] gate pre-acts (64 KB)
  float*    lred = zbuf + 16 * 1024;           // [256]
  _Float16* h16  = (_Float16*)(lred + 256);    // [16][256]
  _Float16* xbuf = h16 + 16 * 256;             // [16][256] = [xh(128) | m(128)]

  const int tid  = threadIdx.x;
  const int r0   = blockIdx.x * 16;            // batch row base
  const int w    = tid >> 5;
  const int lane = tid & 31;
  const int nlo  = lane & 15;
  const int mb   = (lane >> 4) << 3;

  float hreg[16], creg[16];
#pragma unroll
  for (int i = 0; i < 16; ++i) { hreg[i] = 0.f; creg[i] = 0.f; }

  float lacc = 0.f;

  for (int t = 0; t < T_ - 1; ++t) {
    // (a) stage [xh, m] and decayed h (f16) into LDS
#pragma unroll
    for (int i = 0; i < 16; ++i) {
      const int idx = tid + i * 256;           // 0..4095, fixed ownership
      const int row = idx >> 8, col = idx & 255;
      const long src = ((long)(r0 + row) * T_ + t) * F_;
      xbuf[row * 256 + col] =
          (col < 128) ? XH16[src + col] : masks16[src + (col - 128)];
      const float h =
          hreg[i] * gamma[((long)(r0 + row) * T_ + t) * H_ + col];
      hreg[i] = h;
      h16[row * 256 + col] = (_Float16)h;
    }
    __syncthreads();

    // (b) gates: z = [xh,m] @ Wih^T + h @ Whh^T + (b_ih + b_hh)
    //     A tiles hoisted: shared by all 8 n-tiles this wave computes.
    {
      v16h ax[8], ah[8];
#pragma unroll
      for (int kk = 0; kk < 8; ++kk) {
        ax[kk] = ld_a((const _Float16*)xbuf + kk * 32, 256);
        ah[kk] = ld_a((const _Float16*)h16 + kk * 32, 256);
      }
#pragma unroll 1
      for (int j = 0; j < 8; ++j) {
        const int n0 = w * 128 + j * 16;
        v8f acc = {};
#pragma unroll
        for (int kk = 0; kk < K2F / 32; ++kk)   // K = 256 over [xh,m]
          acc = wmma16(ax[kk], ld_b(Wih16 + (long)n0 * K2F + kk * 32, K2F), acc);
#pragma unroll
        for (int kk = 0; kk < H_ / 32; ++kk)    // K = 256 over h
          acc = wmma16(ah[kk], ld_b(Whh16 + (long)n0 * H_ + kk * 32, H_), acc);
        const float bias = b_ih[n0 + nlo] + b_hh[n0 + nlo];
#pragma unroll
        for (int v = 0; v < 8; ++v)
          zbuf[(mb + v) * 1024 + n0 + nlo] = acc[v] + bias;
      }
    }
    __syncthreads();

    // (c) LSTM cell update (torch gate order i,f,g,o)
#pragma unroll
    for (int i = 0; i < 16; ++i) {
      const int idx = tid + i * 256;
      const int row = idx >> 8, col = idx & 255;
      const float ig = sigf(zbuf[row * 1024 + col]);
      const float fg = sigf(zbuf[row * 1024 + 256 + col]);
      const float gg = tanhf(zbuf[row * 1024 + 512 + col]);
      const float og = sigf(zbuf[row * 1024 + 768 + col]);
      const float c  = fg * creg[i] + ig * gg;
      creg[i] = c;
      const float h = og * tanhf(c);
      hreg[i] = h;
      h16[row * 256 + col] = (_Float16)h;
    }
    __syncthreads();

    // (d) out = h @ out_W^T + out_b ; L1 loss vs values[:, t+1]
    {
      const int n0 = w * 16;                   // 8 waves cover F = 128
      v8f acc = {};
#pragma unroll
      for (int kk = 0; kk < H_ / 32; ++kk) {
        v16h a  = ld_a((const _Float16*)h16 + kk * 32, 256);
        v16h wb = ld_b(outW16 + (long)n0 * H_ + kk * 32, H_);
        acc = wmma16(a, wb, acc);
      }
      const float ob  = out_b[n0 + nlo];
      const float inv = invden[t + 1];
#pragma unroll
      for (int v = 0; v < 8; ++v) {
        const int row = mb + v;
        const long idx = ((long)(r0 + row) * T_ + (t + 1)) * F_ + n0 + nlo;
        lacc += fabsf(values[idx] - (acc[v] + ob)) * masks[idx] * inv;
      }
    }

    // prefetch next step activations (gfx1250 global_prefetch_b8)
    if (t + 1 < T_ - 1) {
      const int row = tid & 15;
      __builtin_prefetch(XH16 + ((long)(r0 + row) * T_ + (t + 1)) * F_, 0, 3);
      __builtin_prefetch(masks16 + ((long)(r0 + row) * T_ + (t + 1)) * F_, 0, 3);
    }
    __syncthreads();  // protect h16/xbuf from next iteration's writes
  }

  // final state out
#pragma unroll
  for (int i = 0; i < 16; ++i) {
    const int idx = tid + i * 256;
    const int row = idx >> 8, col = idx & 255;
    h_out[(long)(r0 + row) * H_ + col] = hreg[i];
    c_out[(long)(r0 + row) * H_ + col] = creg[i];
  }
  // block loss partial (deterministic tree)
  lred[tid] = lacc;
  __syncthreads();
  for (int off = 128; off > 0; off >>= 1) {
    if (tid < off) lred[tid] += lred[tid + off];
    __syncthreads();
  }
  if (tid == 0) lossPart[blockIdx.x] = lred[0];
}

// Fixed-order final loss sum (deterministic, no float atomics anywhere)
__global__ void k_loss_final(const float* __restrict__ lossPart,
                             float* __restrict__ loss_out) {
  if (threadIdx.x == 0 && blockIdx.x == 0) {
    float s = 0.f;
    for (int i = 0; i < B_ / 16; ++i) s += lossPart[i];
    *loss_out = s;
  }
}

// ---------------------------------------------------------------------------
extern "C" void kernel_launch(void* const* d_in, const int* in_sizes, int n_in,
                              void* d_out, int out_size, void* d_ws,
                              size_t ws_size, hipStream_t stream) {
  const float* values  = (const float*)d_in[0];
  const float* masks   = (const float*)d_in[1];
  const float* deltas  = (const float*)d_in[2];
  const float* feat_W  = (const float*)d_in[3];
  const float* feat_b  = (const float*)d_in[4];
  const float* temp_W  = (const float*)d_in[5];
  const float* temp_b  = (const float*)d_in[6];
  const float* decay_W = (const float*)d_in[7];
  const float* decay_b = (const float*)d_in[8];
  const float* W_ih    = (const float*)d_in[9];
  const float* W_hh    = (const float*)d_in[10];
  const float* b_ih    = (const float*)d_in[11];
  const float* b_hh    = (const float*)d_in[12];
  const float* out_W   = (const float*)d_in[13];
  const float* out_b   = (const float*)d_in[14];

  float* h_out    = (float*)d_out;
  float* c_out    = h_out + (long)B_ * H_;
  float* loss_out = c_out + (long)B_ * H_;

  // workspace carve (256B aligned)
  char* ws = (char*)d_ws;
  auto alloc = [&](size_t bytes) {
    char* p = ws;
    ws += (bytes + 255) & ~(size_t)255;
    return p;
  };
  _Float16* vals16   = (_Float16*)alloc((size_t)BT * F_ * 2);
  _Float16* valsT16  = (_Float16*)alloc((size_t)BT * F_ * 2);
  _Float16* masks16  = (_Float16*)alloc((size_t)BT * F_ * 2);
  _Float16* deltas16 = (_Float16*)alloc((size_t)BT * F_ * 2);
  _Float16* XH16     = (_Float16*)alloc((size_t)BT * F_ * 2);
  float*    gamma    = (float*)alloc((size_t)BT * H_ * 4);
  _Float16* featW16  = (_Float16*)alloc((size_t)F_ * F_ * 2);
  _Float16* tempW16  = (_Float16*)alloc((size_t)T_ * T_ * 2);
  _Float16* decayW16 = (_Float16*)alloc((size_t)H_ * F_ * 2);
  _Float16* Wih16    = (_Float16*)alloc((size_t)G4H * K2F * 2);
  _Float16* Whh16    = (_Float16*)alloc((size_t)G4H * H_ * 2);
  _Float16* outW16   = (_Float16*)alloc((size_t)F_ * H_ * 2);
  float*    invden   = (float*)alloc((size_t)T_ * 4);
  float*    lossPart = (float*)alloc((size_t)(B_ / 16) * 4);

  const dim3 blk(256);
  const int NE = BT * F_;
  k_cast_f16<<<(NE + 255) / 256, blk, 0, stream>>>(values, vals16, NE);
  k_cast_f16<<<(NE + 255) / 256, blk, 0, stream>>>(masks, masks16, NE);
  k_cast_f16<<<(NE + 255) / 256, blk, 0, stream>>>(deltas, deltas16, NE);
  k_transpose_f16<<<(NE + 255) / 256, blk, 0, stream>>>(values, valsT16);
  k_cast_nodiag<<<(F_ * F_ + 255) / 256, blk, 0, stream>>>(feat_W, featW16, F_);
  k_cast_nodiag<<<(T_ * T_ + 255) / 256, blk, 0, stream>>>(temp_W, tempW16, T_);
  k_cast_f16<<<(H_ * F_ + 255) / 256, blk, 0, stream>>>(decay_W, decayW16, H_ * F_);
  k_cast_f16<<<(G4H * K2F + 255) / 256, blk, 0, stream>>>(W_ih, Wih16, G4H * K2F);
  k_cast_f16<<<(G4H * H_ + 255) / 256, blk, 0, stream>>>(W_hh, Whh16, G4H * H_);
  k_cast_f16<<<(F_ * H_ + 255) / 256, blk, 0, stream>>>(out_W, outW16, F_ * H_);

  k_impute<<<dim3(T_ / 16, B_), blk, 0, stream>>>(
      vals16, valsT16, featW16, tempW16, values, masks, feat_b, temp_b, XH16);
  k_gamma<<<dim3(BT / 16, 2), blk, 0, stream>>>(deltas16, decayW16, decay_b,
                                                gamma);
  k_invden<<<T_ - 1, blk, 0, stream>>>(masks, invden);

  const size_t smem = (size_t)16 * 1024 * 4      // zbuf f32
                    + 256 * 4                    // lred
                    + (size_t)16 * 256 * 2 * 2;  // h16 + xbuf f16
  hipFuncSetAttribute(reinterpret_cast<const void*>(k_scan),
                      hipFuncAttributeMaxDynamicSharedMemorySize, (int)smem);
  k_scan<<<B_ / 16, blk, smem, stream>>>(XH16, masks16, gamma, Wih16, Whh16,
                                         outW16, b_ih, b_hh, out_b, values,
                                         masks, invden, h_out, c_out, lossPart);
  k_loss_final<<<1, 32, 0, stream>>>(lossPart, loss_out);

  (void)in_sizes; (void)n_in; (void)out_size; (void)ws_size;
}
